// ManiPointNet_44169443672115
// MI455X (gfx1250) — matrix operand order repaired and checked by reading, compiled, not run on hardware
//
#include <hip/hip_runtime.h>
#include <cstddef>

// ---------------------------------------------------------------------------
// ManiPointNet forward for MI455X (gfx1250, wave32, WMMA 16x16x32 f16->f32).
//
// Input flattening assumption (JAX pytree order: dict keys sorted, tuples in
// order):
//   0..13   : fc1_b, fc1_w, fc3_b, fc3_w, fc4_b, fc4_w, fc5_b, fc5_w,
//             gn1_b, gn1_g, gn3_b, gn3_g, gn4_b, gn4_g
//   14..45  : sa1   (32 leaves)   46..77 : sa1_g
//   78..109 : sa2               110..141 : sa2_g
//   142..173: sa3               174..205 : sa3_g
//   206     : xyz      (16,4,2048)
//   207     : xyz_goal (16,3,2048)
// Per-SA leaf order (keys sorted: bnl_b,bnl_g,densitynet,lin_b,lin_w,mlp,
// weightnet):
//   +0 bnl_b  +1 bnl_g
//   +2..5   dn0 (w,b,g,be)   +6..9  dn1   +10..13 dn2
//   +14 lin_b  +15 lin_w
//   +16..19 mlp (w,b,g,be)
//   +20..23 wn0  +24..27 wn1  +28..31 wn2
// ---------------------------------------------------------------------------

typedef _Float16 f16_t;
typedef __attribute__((ext_vector_type(16))) _Float16 v16h;
typedef __attribute__((ext_vector_type(8)))  float    v8f;

#define BSZ 16

// LDS tiles are stored in *fragment order*: lane L's 16 halves live at
// [L*16 .. L*16+15], so each lane fetches its whole WMMA operand with two
// ds_load_b128 (32B aligned) instead of 16 scalar ds_load_u16.
//
// Inverse of ISA 7.12.2 16-bit layouts:
//   A (16x32, elem m,k): lane = m + ((k&8)?16:0), slot = (k&7) | ((k&16)>>1)
//   B (32x16, elem k,n): lane = n + ((k&8)?16:0), slot = (k&7) | ((k&16)>>1)
__device__ __forceinline__ int a_slot_idx(int m, int k) {
  return (m + ((k & 8) ? 16 : 0)) * 16 + ((k & 7) | ((k & 16) >> 1));
}
__device__ __forceinline__ int b_slot_idx(int k, int n) {
  return (n + ((k & 8) ? 16 : 0)) * 16 + ((k & 7) | ((k & 16) >> 1));
}

// ------------------------- generic GEMM (WMMA) -----------------------------
// C(M,O) = A(M,K;lda) * W(O,K)^T + bias(O).  One wave per 16x16 C tile.
__global__ void gemm_wmma_kernel(const float* __restrict__ A, int lda,
                                 const float* __restrict__ W,
                                 const float* __restrict__ bias,
                                 float* __restrict__ C, int ldc,
                                 int M, int K, int O) {
  __shared__ __attribute__((aligned(32))) f16_t sA[16 * 32];
  __shared__ __attribute__((aligned(32))) f16_t sB[32 * 16];
  int m0 = blockIdx.x * 16, o0 = blockIdx.y * 16;
  int lane = threadIdx.x;
  v8f acc = {};
  for (int k0 = 0; k0 < K; k0 += 32) {
    if (k0 + 32 < K)  // hint next K tile (global_prefetch_b8)
      __builtin_prefetch(A + (size_t)(m0 + (lane & 15)) * lda + k0 + 32, 0, 1);
    for (int i = lane; i < 512; i += 32) {
      int r = i >> 5, c = i & 31;
      float va = (m0 + r < M && k0 + c < K) ? A[(size_t)(m0 + r) * lda + k0 + c] : 0.f;
      sA[a_slot_idx(r, c)] = (f16_t)va;
      int n = i & 15, kk = i >> 4;
      float vb = (o0 + n < O && k0 + kk < K) ? W[(size_t)(o0 + n) * K + k0 + kk] : 0.f;
      sB[b_slot_idx(kk, n)] = (f16_t)vb;
    }
    __syncthreads();
    v16h af = *reinterpret_cast<const v16h*>(sA + lane * 16);
    v16h bf = *reinterpret_cast<const v16h*>(sB + lane * 16);
    acc = __builtin_amdgcn_wmma_f32_16x16x32_f16(false, af, false, bf,
                                                 (short)0, acc, false, false);
    __syncthreads();
  }
  int col = lane & 15, rbase = (lane >> 4) * 8;
#pragma unroll
  for (int g = 0; g < 8; ++g) {
    int r = m0 + rbase + g, c = o0 + col;
    if (r < M && c < O)
      C[(size_t)r * ldc + c] = acc[g] + (bias ? bias[c] : 0.f);
  }
}

// ------------- per-site einsum: out[site,c,w] = sum_k feat*dsc * wgt --------
// feat:(S,k,Cf) f32, wgt:(S,k,16) f32, dsc:(S*k) f32, out:(S,Cf,16) f32.
__global__ void site_einsum_kernel(const float* __restrict__ feat,
                                   const float* __restrict__ wgt,
                                   const float* __restrict__ dsc,
                                   float* __restrict__ out,
                                   int S, int ksamp, int Cf) {
  __shared__ __attribute__((aligned(32))) f16_t sA[16 * 32];
  __shared__ __attribute__((aligned(32))) f16_t sB[32 * 16];
  int site = blockIdx.x, ctile = blockIdx.y, lane = threadIdx.x;
  const float* fp = feat + (size_t)site * ksamp * Cf;
  const float* wp = wgt + (size_t)site * ksamp * 16;
  const float* dp = dsc + (size_t)site * ksamp;
  v8f acc = {};
  for (int k0 = 0; k0 < ksamp; k0 += 32) {
    for (int i = lane; i < 512; i += 32) {
      int r = i >> 5, kk = i & 31;  // A = feat^T : rows c, cols k
      int c = ctile * 16 + r, ks = k0 + kk;
      float va = (ks < ksamp && c < Cf) ? fp[(size_t)ks * Cf + c] * dp[ks] : 0.f;
      sA[a_slot_idx(r, kk)] = (f16_t)va;
      int n = i & 15, kb = i >> 4;
      float vb = (k0 + kb < ksamp) ? wp[(size_t)(k0 + kb) * 16 + n] : 0.f;
      sB[b_slot_idx(kb, n)] = (f16_t)vb;
    }
    __syncthreads();
    v16h af = *reinterpret_cast<const v16h*>(sA + lane * 16);
    v16h bf = *reinterpret_cast<const v16h*>(sB + lane * 16);
    acc = __builtin_amdgcn_wmma_f32_16x16x32_f16(false, af, false, bf,
                                                 (short)0, acc, false, false);
    __syncthreads();
  }
  int col = lane & 15, rbase = (lane >> 4) * 8;
#pragma unroll
  for (int g = 0; g < 8; ++g) {
    int c = ctile * 16 + rbase + g;
    if (c < Cf) out[((size_t)site * Cf + c) * 16 + col] = acc[g];
  }
}

// ----------------------------- BN (train mode) -----------------------------
__global__ void bn_stats_kernel(const float* __restrict__ Y, int ldy,
                                int M, int O, float* mean, float* rstd) {
  int o = blockIdx.x;
  __shared__ float s1[256], s2[256];
  float a = 0.f, b = 0.f;
  for (int r = threadIdx.x; r < M; r += 256) {
    float v = Y[(size_t)r * ldy + o];
    a += v; b += v * v;
  }
  s1[threadIdx.x] = a; s2[threadIdx.x] = b;
  __syncthreads();
  for (int st = 128; st > 0; st >>= 1) {
    if ((int)threadIdx.x < st) { s1[threadIdx.x] += s1[threadIdx.x + st];
                                 s2[threadIdx.x] += s2[threadIdx.x + st]; }
    __syncthreads();
  }
  if (threadIdx.x == 0) {
    float m = s1[0] / (float)M;
    float v = s2[0] / (float)M - m * m;
    mean[o] = m; rstd[o] = rsqrtf(v + 1e-5f);
  }
}

__global__ void bn_apply_relu_kernel(const float* __restrict__ Y, int ldy,
                                     int M, int O,
                                     const float* mean, const float* rstd,
                                     const float* g, const float* b,
                                     float* __restrict__ out, int ldo) {
  size_t i = (size_t)blockIdx.x * blockDim.x + threadIdx.x;
  size_t tot = (size_t)M * O;
  if (i >= tot) return;
  int r = (int)(i / O), o = (int)(i % O);
  float v = (Y[(size_t)r * ldy + o] - mean[o]) * rstd[o] * g[o] + b[o];
  out[(size_t)r * ldo + o] = fmaxf(v, 0.f);
}

// ------------------------------ layernorm (_gn) ----------------------------
__global__ void layernorm_relu_kernel(const float* __restrict__ X, int D,
                                      const float* g, const float* b,
                                      float* __restrict__ out) {
  int r = blockIdx.x, lane = threadIdx.x;
  float s = 0.f, s2 = 0.f;
  for (int i = lane; i < D; i += 32) {
    float v = X[(size_t)r * D + i];
    s += v; s2 += v * v;
  }
  for (int o = 16; o > 0; o >>= 1) { s += __shfl_xor(s, o, 32); s2 += __shfl_xor(s2, o, 32); }
  float m = s / (float)D;
  float rs = rsqrtf(s2 / (float)D - m * m + 1e-5f);
  for (int i = lane; i < D; i += 32) {
    float v = (X[(size_t)r * D + i] - m) * rs * g[i] + b[i];
    out[(size_t)r * D + i] = fmaxf(v, 0.f);
  }
}

// ------------------------------ transpose ----------------------------------
__global__ void transpose_bcn_kernel(const float* __restrict__ in,
                                     float* __restrict__ out, int C, int N) {
  int i = blockIdx.x * blockDim.x + threadIdx.x;
  int tot = BSZ * N;
  if (i >= tot) return;
  int b = i / N, n = i % N;
  for (int c = 0; c < C; ++c)
    out[((size_t)b * N + n) * C + c] = in[((size_t)b * C + c) * N + n];
}

// -------------------------- density (fused N^2) ----------------------------
__global__ void density_kernel(const float* __restrict__ xyz, int cs, int N,
                               float bw, float* __restrict__ invDen) {
  int b = blockIdx.y;
  int n = blockIdx.x * 256 + threadIdx.x;
  __shared__ float sx[256][3];
  const float* base = xyz + (size_t)b * N * cs;
  float px = 0.f, py = 0.f, pz = 0.f;
  if (n < N) { const float* p = base + (size_t)n * cs; px = p[0]; py = p[1]; pz = p[2]; }
  float acc = 0.f;
  float c0 = -1.f / (2.f * bw * bw);
  for (int m0 = 0; m0 < N; m0 += 256) {
    int mi = m0 + threadIdx.x;
    if (mi < N) {
      const float* p = base + (size_t)mi * cs;
      sx[threadIdx.x][0] = p[0]; sx[threadIdx.x][1] = p[1]; sx[threadIdx.x][2] = p[2];
    }
    __syncthreads();
    int lim = N - m0; if (lim > 256) lim = 256;
    for (int j = 0; j < lim; ++j) {
      float dx = px - sx[j][0], dy = py - sx[j][1], dz = pz - sx[j][2];
      acc += expf(c0 * (dx * dx + dy * dy + dz * dz));
    }
    __syncthreads();
  }
  if (n < N) invDen[(size_t)b * N + n] = (2.5f * bw * (float)N) / acc;
}

// ----------------------- farthest point sampling ---------------------------
__global__ void fps_kernel(const float* __restrict__ xyz, int cs, int N,
                           int np, int* __restrict__ cent,
                           float* __restrict__ nxyz) {
  int b = blockIdx.x, t = threadIdx.x;
  extern __shared__ float dist[];
  __shared__ float rv[256]; __shared__ int ri[256]; __shared__ int curF;
  const float* base = xyz + (size_t)b * N * cs;
  for (int i = t; i < N; i += 256) dist[i] = 1e10f;
  if (t == 0) curF = 0;
  __syncthreads();
  for (int it = 0; it < np; ++it) {
    int far = curF;
    const float* fp = base + (size_t)far * cs;
    float cx = fp[0], cy = fp[1], cz = fp[2];
    if (t == 0) {
      cent[(size_t)b * np + it] = far;
      float* o = nxyz + ((size_t)b * np + it) * 3;
      o[0] = cx; o[1] = cy; o[2] = cz;
    }
    float best = -1.f; int bi = 0;
    for (int i = t; i < N; i += 256) {
      const float* p = base + (size_t)i * cs;
      float dx = p[0] - cx, dy = p[1] - cy, dz = p[2] - cz;
      float dm = fminf(dist[i], dx * dx + dy * dy + dz * dz);
      dist[i] = dm;
      if (dm > best) { best = dm; bi = i; }
    }
    rv[t] = best; ri[t] = bi;
    __syncthreads();
    for (int st = 128; st > 0; st >>= 1) {
      if (t < st) {
        if (rv[t + st] > rv[t] || (rv[t + st] == rv[t] && ri[t + st] < ri[t])) {
          rv[t] = rv[t + st]; ri[t] = ri[t + st];
        }
      }
      __syncthreads();
    }
    if (t == 0) curF = ri[0];
    __syncthreads();
  }
}

// ----------------------------- kNN (top_k order) ---------------------------
__global__ void knn_kernel(const float* __restrict__ nxyz,
                           const float* __restrict__ xyz, int cs, int N,
                           int Bs, int s, int k, int* __restrict__ idxOut) {
  int gid = blockIdx.x * blockDim.x + threadIdx.x;
  if (gid >= Bs) return;
  int b = gid / s;
  const float* base = xyz + (size_t)b * N * cs;
  float qx = nxyz[(size_t)gid * 3], qy = nxyz[(size_t)gid * 3 + 1],
        qz = nxyz[(size_t)gid * 3 + 2];
  float lastd = -3.4e38f; int lasti = -1;
  for (int j = 0; j < k; ++j) {
    float bd = 3.4e38f; int bi = 0;
    for (int i = 0; i < N; ++i) {
      const float* p = base + (size_t)i * cs;
      float dx = p[0] - qx, dy = p[1] - qy, dz = p[2] - qz;
      float d = dx * dx + dy * dy + dz * dz;
      bool gt = (d > lastd) || (d == lastd && i > lasti);
      bool lt = (d < bd) || (d == bd && i < bi);
      if (gt && lt) { bd = d; bi = i; }
    }
    idxOut[(size_t)gid * k + j] = bi;
    lastd = bd; lasti = bi;
  }
}

// --------------------- grouping / gather / density scale -------------------
__global__ void group_kernel(const float* __restrict__ xyz, int cs,
                             const float* __restrict__ pts, int C,
                             const float* __restrict__ invDen,
                             const float* __restrict__ nxyz,
                             const int* __restrict__ idx,
                             int N, int s, int k,
                             float* __restrict__ gxyz, float* __restrict__ npts,
                             int inC, float* __restrict__ ds0) {
  int site = blockIdx.x, b = site / s, t = threadIdx.x;
  __shared__ float red[128];
  int ii = idx ? idx[(size_t)site * k + t] : t;
  const float* p = xyz + ((size_t)b * N + ii) * cs;
  float ox = 0.f, oy = 0.f, oz = 0.f;
  if (nxyz) { ox = nxyz[(size_t)site * 3]; oy = nxyz[(size_t)site * 3 + 1];
              oz = nxyz[(size_t)site * 3 + 2]; }
  float gx = p[0] - ox, gy = p[1] - oy, gz = p[2] - oz;
  size_t row = (size_t)site * k + t;
  gxyz[row * 3] = gx; gxyz[row * 3 + 1] = gy; gxyz[row * 3 + 2] = gz;
  float* d = npts + row * inC;
  d[0] = gx; d[1] = gy; d[2] = gz;
  const float* pp = pts + ((size_t)b * N + ii) * C;
  for (int c = 0; c < C; ++c) d[3 + c] = pp[c];
  float g = invDen[(size_t)b * N + ii];
  red[t] = g;
  __syncthreads();
  for (int st = k >> 1; st > 0; st >>= 1) {
    if (t < st) red[t] = fmaxf(red[t], red[t + st]);
    __syncthreads();
  }
  ds0[row] = g / red[0];
}

// ------------------------------ log_softmax --------------------------------
__global__ void logsoftmax2_kernel(const float* __restrict__ lg,
                                   float* __restrict__ out, int Bn) {
  int b = threadIdx.x;
  if (b >= Bn) return;
  float a = lg[b * 2], c = lg[b * 2 + 1];
  float m = fmaxf(a, c);
  float lse = m + logf(expf(a - m) + expf(c - m));
  out[b * 2] = a - lse;
  out[b * 2 + 1] = c - lse;
}

// =============================== host side =================================
namespace {

struct Alloc {
  char* base; size_t off; size_t cap;
  void* get(size_t bytes) {
    size_t o = (off + 255) & ~(size_t)255;
    off = o + bytes;
    return base + o;
  }
  float* f(size_t n) { return (float*)get(n * sizeof(float)); }
  int*   i(size_t n) { return (int*)get(n * sizeof(int)); }
};

inline const float* P(void* const* din, int i) { return (const float*)din[i]; }

void gemm(const float* A, int lda, const float* W, const float* bias,
          float* C, int ldc, int M, int K, int O, hipStream_t st) {
  dim3 g((M + 15) / 16, (O + 15) / 16);
  gemm_wmma_kernel<<<g, 32, 0, st>>>(A, lda, W, bias, C, ldc, M, K, O);
}

// einsum('...c,oc->...o') + bias, train-mode BN over all rows, ReLU. In place.
float* mlp_layer(void* const* din, const float* A, int M, int K,
                 int wI, int bI, int gI, int beI, int O,
                 Alloc& al, hipStream_t st) {
  float* y = al.f((size_t)M * O);
  float* mn = al.f(O);
  float* rs = al.f(O);
  gemm(A, K, P(din, wI), P(din, bI), y, O, M, K, O, st);
  bn_stats_kernel<<<O, 256, 0, st>>>(y, O, M, O, mn, rs);
  size_t tot = (size_t)M * O;
  bn_apply_relu_kernel<<<(unsigned)((tot + 255) / 256), 256, 0, st>>>(
      y, O, M, O, mn, rs, P(din, gI), P(din, beI), y, O);
  return y;
}

void run_sa(void* const* din, int p0, const float* xyz, int cs,
            const float* pts, int C, int N, int s, int k, float bw, int cout,
            float* newxyz, float* outFeat, int ldo, bool groupAll,
            Alloc& al, hipStream_t st) {
  int inC = 3 + C;
  int M = BSZ * s * k;
  float* invDen = al.f((size_t)BSZ * N);
  density_kernel<<<dim3((N + 255) / 256, BSZ), 256, 0, st>>>(xyz, cs, N, bw, invDen);
  int* knnIdx = nullptr;
  if (!groupAll) {
    int* fpsIdx = al.i((size_t)BSZ * s);
    fps_kernel<<<BSZ, 256, (size_t)N * sizeof(float), st>>>(xyz, cs, N, s, fpsIdx, newxyz);
    knnIdx = al.i((size_t)M);
    int Bs = BSZ * s;
    knn_kernel<<<(Bs + 63) / 64, 64, 0, st>>>(newxyz, xyz, cs, N, Bs, s, k, knnIdx);
  }
  float* gxyz = al.f((size_t)M * 3);
  float* npts = al.f((size_t)M * inC);
  float* ds0  = al.f((size_t)M);
  group_kernel<<<BSZ * s, k, 0, st>>>(xyz, cs, pts, C, invDen,
                                      groupAll ? nullptr : newxyz, knnIdx,
                                      N, s, k, gxyz, npts, inC, ds0);
  // feature MLP (one layer)
  float* feat = mlp_layer(din, npts, M, inC, p0 + 16, p0 + 17, p0 + 18, p0 + 19, cout, al, st);
  // densitynet 1 -> 16 -> 8 -> 1
  float* d1 = mlp_layer(din, ds0, M, 1,  p0 + 2,  p0 + 3,  p0 + 4,  p0 + 5,  16, al, st);
  float* d2 = mlp_layer(din, d1,  M, 16, p0 + 6,  p0 + 7,  p0 + 8,  p0 + 9,  8,  al, st);
  float* dsc= mlp_layer(din, d2,  M, 8,  p0 + 10, p0 + 11, p0 + 12, p0 + 13, 1,  al, st);
  // weightnet 3 -> 8 -> 8 -> 16
  float* w1 = mlp_layer(din, gxyz, M, 3, p0 + 20, p0 + 21, p0 + 22, p0 + 23, 8,  al, st);
  float* w2 = mlp_layer(din, w1,   M, 8, p0 + 24, p0 + 25, p0 + 26, p0 + 27, 8,  al, st);
  float* wg = mlp_layer(din, w2,   M, 8, p0 + 28, p0 + 29, p0 + 30, p0 + 31, 16, al, st);
  // per-site einsum -> (B*s, cout*16)
  float* eo = al.f((size_t)BSZ * s * cout * 16);
  site_einsum_kernel<<<dim3(BSZ * s, cout / 16), 32, 0, st>>>(feat, wg, dsc, eo, BSZ * s, k, cout);
  // linear + BN(axes 0,1) + ReLU
  int Ms = BSZ * s;
  float* ly = al.f((size_t)Ms * cout);
  float* mn = al.f(cout);
  float* rs = al.f(cout);
  gemm(eo, 16 * cout, P(din, p0 + 15), P(din, p0 + 14), ly, cout, Ms, 16 * cout, cout, st);
  bn_stats_kernel<<<cout, 256, 0, st>>>(ly, cout, Ms, cout, mn, rs);
  size_t tot = (size_t)Ms * cout;
  bn_apply_relu_kernel<<<(unsigned)((tot + 255) / 256), 256, 0, st>>>(
      ly, cout, Ms, cout, mn, rs, P(din, p0 + 1), P(din, p0 + 0), outFeat, ldo);
}

void run_branch(void* const* din, const float* raw, int C, const int* saB,
                float* headCol, Alloc& al, hipStream_t st) {
  const int N0 = 2048;
  float* lT = al.f((size_t)BSZ * N0 * C);
  transpose_bcn_kernel<<<(BSZ * N0 + 255) / 256, 256, 0, st>>>(raw, lT, C, N0);
  float* x1 = al.f((size_t)BSZ * 512 * 3);
  float* f1 = al.f((size_t)BSZ * 512 * 64);
  float* x2 = al.f((size_t)BSZ * 128 * 3);
  float* f2 = al.f((size_t)BSZ * 128 * 128);
  size_t mark = al.off;
  run_sa(din, saB[0], lT, C, lT, C,   2048, 512, 32,  0.1f, 64,  x1, f1, 64,  false, al, st);
  al.off = mark;
  run_sa(din, saB[1], x1, 3, f1, 64,  512,  128, 64,  0.2f, 128, x2, f2, 128, false, al, st);
  al.off = mark;
  run_sa(din, saB[2], x2, 3, f2, 128, 128,  1,   128, 0.4f, 256, nullptr, headCol, 512, true, al, st);
  al.off = mark;
}

}  // namespace

extern "C" void kernel_launch(void* const* d_in, const int* in_sizes, int n_in,
                              void* d_out, int out_size, void* d_ws, size_t ws_size,
                              hipStream_t stream) {
  (void)in_sizes; (void)n_in; (void)out_size;
  Alloc al{(char*)d_ws, 0, ws_size};
  float* head = al.f((size_t)BSZ * 512);  // [f3 | h3] concat, 16 x 512

  static const int saA[3] = {14, 78, 142};
  static const int saG[3] = {46, 110, 174};
  size_t mark = al.off;
  run_branch(d_in, (const float*)d_in[206], 4, saA, head,       al, stream);
  al.off = mark;
  run_branch(d_in, (const float*)d_in[207], 3, saG, head + 256, al, stream);
  al.off = mark;

  // FC head: fc1 -> gn -> relu -> fc3 -> gn -> relu -> fc4 -> gn -> relu -> fc5
  float* t1 = al.f((size_t)BSZ * 256);
  gemm(head, 512, P(d_in, 1), P(d_in, 0), t1, 256, BSZ, 512, 256, stream);
  layernorm_relu_kernel<<<BSZ, 32, 0, stream>>>(t1, 256, P(d_in, 9), P(d_in, 8), t1);
  float* t2 = al.f((size_t)BSZ * 128);
  gemm(t1, 256, P(d_in, 3), P(d_in, 2), t2, 128, BSZ, 256, 128, stream);
  layernorm_relu_kernel<<<BSZ, 32, 0, stream>>>(t2, 128, P(d_in, 11), P(d_in, 10), t2);
  float* t3 = al.f((size_t)BSZ * 64);
  gemm(t2, 128, P(d_in, 5), P(d_in, 4), t3, 64, BSZ, 128, 64, stream);
  layernorm_relu_kernel<<<BSZ, 32, 0, stream>>>(t3, 64, P(d_in, 13), P(d_in, 12), t3);
  float* lg = al.f((size_t)BSZ * 2);
  gemm(t3, 64, P(d_in, 7), P(d_in, 6), lg, 2, BSZ, 64, 2, stream);
  logsoftmax2_kernel<<<1, 32, 0, stream>>>(lg, (float*)d_out, BSZ);
}
